// TransformerConvBlock_31198642438245
// MI455X (gfx1250) — compile-verified, matrix operand order checked by reading, not run on hardware
//
#include <hip/hip_runtime.h>

typedef _Float16 h16;
typedef __attribute__((ext_vector_type(8)))  _Float16 v8h;
typedef __attribute__((ext_vector_type(16))) _Float16 v16h;
typedef __attribute__((ext_vector_type(8)))  float    v8f;
typedef __attribute__((ext_vector_type(4)))  int      v4i;

#define HW     16384
#define TILE   16
#define WAVES  4
#define BLKPIX (WAVES * TILE)

#if defined(__has_builtin)
#if __has_builtin(__builtin_amdgcn_global_load_async_to_lds_b128)
#define HAVE_ASYNC_B128 1
#endif
#if __has_builtin(__builtin_amdgcn_s_wait_asynccnt)
#define HAVE_WAIT_ASYNC 1
#endif
#endif
#ifndef HAVE_ASYNC_B128
#define HAVE_ASYNC_B128 0
#endif
#ifndef HAVE_WAIT_ASYNC
#define HAVE_WAIT_ASYNC 0
#endif
#define USE_ASYNC_COPY (HAVE_ASYNC_B128 && HAVE_WAIT_ASYNC)

union U8f { v8f v; float f[8]; };
union U8h { v8h v; h16  f[8]; };

static __device__ __forceinline__ v16h cat16(v8h a, v8h b) {
  return __builtin_shufflevector(a, b, 0,1,2,3,4,5,6,7,8,9,10,11,12,13,14,15);
}

static __device__ __forceinline__ v8h lds8(const h16* p) {
  return *(const v8h*)p;
}

static __device__ __forceinline__ v8f wmma_f16(v16h a, v16h b, v8f c) {
  // D = A(16x32 f16) x B(32x16 f16) + C(16x16 f32)
  return __builtin_amdgcn_wmma_f32_16x16x32_f16(false, a, false, b, (short)0, c,
                                                false, false);
}

// ---------------------------------------------------------------------------
// Prep: convert the six 64x64 f32 weight matrices into f16 B-fragment layout.
// Fragment gf = g*8 + nt*2 + kf. Lane L<16 holds col n = nt*16+L with
// K {kb..kb+7, kb+16..kb+23}, kb = kf*32 (+8 for lanes 16..31).
// g<4 (Wq,Wk,Wv,Wo):  B[k][n] = Wflat[k*64+n]   (x @ W)
// g>=4 (ffn1,ffn2):   B[k][n] = Wflat[n*64+k]   (y @ W.T)
// ---------------------------------------------------------------------------
__global__ void tcb_prep_weights(const float* __restrict__ Wq, const float* __restrict__ Wk,
                                 const float* __restrict__ Wv, const float* __restrict__ Wo,
                                 const float* __restrict__ F1, const float* __restrict__ F2,
                                 h16* __restrict__ out) {
  const int t = blockIdx.x * 256 + threadIdx.x;   // 0..24575
  const int g = t >> 12;
  const int r = t & 4095;
  const int f = r >> 9;            // fragment-in-matrix 0..7
  const int L = (r >> 4) & 31;     // lane
  const int i = r & 15;            // half index within lane
  const int nt = f >> 1;
  const int kf = f & 1;
  const int n  = nt * 16 + (L & 15);
  const int kb = kf * 32 + ((L >> 4) << 3);
  const int k  = kb + ((i < 8) ? i : (8 + i));    // i>=8 -> kb+16+(i-8)
  const float* W = (g == 0) ? Wq : (g == 1) ? Wk : (g == 2) ? Wv
                 : (g == 3) ? Wo : (g == 4) ? F1 : F2;
  const float v = (g < 4) ? W[k * 64 + n] : W[n * 64 + k];
  out[t] = (h16)v;
}

// ---------------------------------------------------------------------------
// Fused transformer block. One wave = 16 consecutive pixels (same b, same row).
// LDS: [0 .. 24575]  weight fragments (f16, 48KB)
//      [24576 ...]   per-wave activation tiles: 16 rows x 256 halfs
//         row layout: [0..63]=xn/y  [64..127]=Q/a  [128..191]=K/y1
//                     [192..255]=V ; [128..255] reused as y2 (f32, 64 floats)
// ---------------------------------------------------------------------------
__global__ __launch_bounds__(128) void tcb_fused(
    const float* __restrict__ x, const h16* __restrict__ wsc,
    const float* __restrict__ g1, const float* __restrict__ b1,
    const float* __restrict__ g2, const float* __restrict__ b2,
    const float* __restrict__ fb1, const float* __restrict__ fb2,
    float* __restrict__ out) {
  extern __shared__ __align__(16) h16 smem[];
  h16* wfrag = smem;              // 24576 halfs
  h16* actb  = smem + 24576;      // WAVES * 4096 halfs

  const int tid  = threadIdx.x;
  const int lane = tid & 31;
  const int wv   = tid >> 5;

  // Weight-fragment fill (48KB global -> LDS). Prefer the CDNA5 async path
  // (GLOBAL_LOAD_ASYNC_TO_LDS_B128, ASYNCcnt) so the copy overlaps Stage 1.
#if USE_ASYNC_COPY
  {
    typedef __attribute__((address_space(1))) v4i gv4i;
    typedef __attribute__((address_space(3))) v4i lv4i;
    gv4i* gp = ((gv4i*)(v4i*)wsc)   + tid;
    lv4i* lp = ((lv4i*)(v4i*)wfrag) + tid;
    #pragma unroll
    for (int i = 0; i < 24; ++i) {
      __builtin_amdgcn_global_load_async_to_lds_b128(gp, lp, 0, 0);
      gp += 128;   // advance 2048 bytes
      lp += 128;
    }
  }
#else
  {
    const uint4* s = (const uint4*)wsc;
    uint4* d = (uint4*)wfrag;
    #pragma unroll
    for (int i = 0; i < 24; ++i) d[tid + 128 * i] = s[tid + 128 * i];
  }
#endif

  h16* act = actb + (wv << 12);
  const int tileP = blockIdx.x * BLKPIX + wv * TILE;
  const int bb = tileP >> 14;          // batch index
  const int hw = tileP & 16383;        // pixel offset within image
  const float* xb = x + ((size_t)bb << 20) + hw;

  const int p  = lane & 15;            // pixel-in-tile / WMMA row-or-col index
  const int hf = lane >> 4;            // half-wave select
  const int c0 = hf << 5;              // this lane's channel base (LN stages)
  const int kb = hf << 3;              // A/B fragment K sub-offset
  const int mh = hf << 3;              // C fragment row offset (+8 for hi lanes)

  // ---- Stage 1: load x (kept in regs for residual) + LayerNorm1 -> xn (f16)
  // x is streamed exactly once -> non-temporal so it doesn't evict L2.
  float xv[32];
  {
    float s = 0.f, s2 = 0.f;
    #pragma unroll
    for (int i = 0; i < 32; ++i) {
      const float v = __builtin_nontemporal_load(xb + (size_t)(c0 + i) * HW + p);
      xv[i] = v; s += v; s2 += v * v;
    }
    s  += __shfl_xor(s, 16);
    s2 += __shfl_xor(s2, 16);
    const float mean = s * (1.f / 64.f);
    const float rstd = rsqrtf(s2 * (1.f / 64.f) - mean * mean + 1e-5f);
    #pragma unroll
    for (int i = 0; i < 32; ++i) {
      const int c = c0 + i;
      act[p * 256 + c] = (h16)((xv[i] - mean) * rstd * g1[c] + b1[c]);
    }
  }
#if USE_ASYNC_COPY
  __builtin_amdgcn_s_wait_asynccnt(0);
#endif
  __syncthreads();

  // ---- Stage 2: Q/K/V = xn @ W  (24 WMMAs). 1/sqrt(8) folded into Q.
  {
    const h16* rowp = act + p * 256;
    const v16h A0 = cat16(lds8(rowp + kb),      lds8(rowp + kb + 16));
    const v16h A1 = cat16(lds8(rowp + kb + 32), lds8(rowp + kb + 48));
    #pragma unroll
    for (int mat = 0; mat < 3; ++mat) {
      #pragma unroll
      for (int nt = 0; nt < 4; ++nt) {
        U8f acc;
        #pragma unroll
        for (int j = 0; j < 8; ++j) acc.f[j] = 0.f;
        const h16* bq = wfrag + (((mat * 8 + nt * 2) * 32 + lane) << 4);
        acc.v = wmma_f16(A0, cat16(lds8(bq),       lds8(bq + 8)),   acc.v);
        acc.v = wmma_f16(A1, cat16(lds8(bq + 512), lds8(bq + 520)), acc.v);
        const float sc = (mat == 0) ? 0.35355339059327373f : 1.f;
        #pragma unroll
        for (int r = 0; r < 8; ++r)
          act[(r + mh) * 256 + 64 + mat * 64 + nt * 16 + p] = (h16)(acc.f[r] * sc);
      }
    }
  }
  __syncthreads();

  // ---- Stage 3: per-pixel head-mixing attention (2 lanes/pixel, 4 heads each)
  {
    const h16* Qr = act + p * 256 + 64;
    const h16* Kr = act + p * 256 + 128;
    const h16* Vr = act + p * 256 + 192;
    U8h Kh[8], Vh[8];
    #pragma unroll
    for (int m = 0; m < 8; ++m) { Kh[m].v = lds8(Kr + m * 8); Vh[m].v = lds8(Vr + m * 8); }
    float aout[4][8];
    #pragma unroll
    for (int n = 0; n < 4; ++n) {
      U8h qh; qh.v = lds8(Qr + (hf * 4 + n) * 8);
      float q[8];
      #pragma unroll
      for (int d = 0; d < 8; ++d) q[d] = (float)qh.f[d];
      float scv[8]; float mx = -3.0e38f;
      #pragma unroll
      for (int m = 0; m < 8; ++m) {
        float sv = 0.f;
        #pragma unroll
        for (int d = 0; d < 8; ++d) sv += q[d] * (float)Kh[m].f[d];
        scv[m] = sv; mx = fmaxf(mx, sv);
      }
      float ssum = 0.f;
      #pragma unroll
      for (int m = 0; m < 8; ++m) { scv[m] = __expf(scv[m] - mx); ssum += scv[m]; }
      const float inv = 1.f / ssum;
      #pragma unroll
      for (int d = 0; d < 8; ++d) {
        float av = 0.f;
        #pragma unroll
        for (int m = 0; m < 8; ++m) av += scv[m] * (float)Vh[m].f[d];
        aout[n][d] = av * inv;
      }
    }
    #pragma unroll
    for (int n = 0; n < 4; ++n)
      #pragma unroll
      for (int d = 0; d < 8; ++d)
        act[p * 256 + 64 + (hf * 4 + n) * 8 + d] = (h16)aout[n][d];  // a over Q
  }
  __syncthreads();

  // ---- Stage 4: x_attn = a @ W_O ; y = relu(xn + x_attn) over xn region
  {
    const h16* rowp = act + p * 256 + 64;
    const v16h A0 = cat16(lds8(rowp + kb),      lds8(rowp + kb + 16));
    const v16h A1 = cat16(lds8(rowp + kb + 32), lds8(rowp + kb + 48));
    #pragma unroll
    for (int nt = 0; nt < 4; ++nt) {
      U8f acc;
      #pragma unroll
      for (int j = 0; j < 8; ++j) acc.f[j] = 0.f;
      const h16* bq = wfrag + (((24 + nt * 2) * 32 + lane) << 4);
      acc.v = wmma_f16(A0, cat16(lds8(bq),       lds8(bq + 8)),   acc.v);
      acc.v = wmma_f16(A1, cat16(lds8(bq + 512), lds8(bq + 520)), acc.v);
      #pragma unroll
      for (int r = 0; r < 8; ++r) {
        const int idx = (r + mh) * 256 + nt * 16 + p;
        act[idx] = (h16)fmaxf((float)act[idx] + acc.f[r], 0.f);
      }
    }
  }
  __syncthreads();

  // ---- Stage 5: y1 = relu(y @ ffn1_w.T + b1) into K region
  {
    const h16* rowp = act + p * 256;
    const v16h A0 = cat16(lds8(rowp + kb),      lds8(rowp + kb + 16));
    const v16h A1 = cat16(lds8(rowp + kb + 32), lds8(rowp + kb + 48));
    #pragma unroll
    for (int nt = 0; nt < 4; ++nt) {
      const float bv = fb1[nt * 16 + p];
      U8f acc;
      #pragma unroll
      for (int j = 0; j < 8; ++j) acc.f[j] = bv;
      const h16* bq = wfrag + (((32 + nt * 2) * 32 + lane) << 4);
      acc.v = wmma_f16(A0, cat16(lds8(bq),       lds8(bq + 8)),   acc.v);
      acc.v = wmma_f16(A1, cat16(lds8(bq + 512), lds8(bq + 520)), acc.v);
      #pragma unroll
      for (int r = 0; r < 8; ++r)
        act[(r + mh) * 256 + 128 + nt * 16 + p] = (h16)fmaxf(acc.f[r], 0.f);
    }
  }
  __syncthreads();

  // ---- Stage 6: y2 = y1 @ ffn2_w.T + b2 (f32 over K/V region)
  {
    const h16* rowp = act + p * 256 + 128;
    const v16h A0 = cat16(lds8(rowp + kb),      lds8(rowp + kb + 16));
    const v16h A1 = cat16(lds8(rowp + kb + 32), lds8(rowp + kb + 48));
    #pragma unroll
    for (int nt = 0; nt < 4; ++nt) {
      const float bv = fb2[nt * 16 + p];
      U8f acc;
      #pragma unroll
      for (int j = 0; j < 8; ++j) acc.f[j] = bv;
      const h16* bq = wfrag + (((40 + nt * 2) * 32 + lane) << 4);
      acc.v = wmma_f16(A0, cat16(lds8(bq),       lds8(bq + 8)),   acc.v);
      acc.v = wmma_f16(A1, cat16(lds8(bq + 512), lds8(bq + 520)), acc.v);
      #pragma unroll
      for (int r = 0; r < 8; ++r)
        *(float*)(act + (r + mh) * 256 + 128 + 2 * (nt * 16 + p)) = acc.f[r];
    }
  }
  __syncthreads();

  // ---- Stage 7: LayerNorm2, identity residual (regs), relu, store NCHW
  // out is written exactly once -> non-temporal store.
  {
    const h16* yrow = act + p * 256 + 128;
    float ys[32]; float s = 0.f, s2 = 0.f;
    #pragma unroll
    for (int i = 0; i < 32; ++i) {
      const float v = *(const float*)(yrow + 2 * (c0 + i));
      ys[i] = v; s += v; s2 += v * v;
    }
    s  += __shfl_xor(s, 16);
    s2 += __shfl_xor(s2, 16);
    const float mean = s * (1.f / 64.f);
    const float rstd = rsqrtf(s2 * (1.f / 64.f) - mean * mean + 1e-5f);
    float* ob = out + ((size_t)bb << 20) + hw;
    #pragma unroll
    for (int i = 0; i < 32; ++i) {
      const int c = c0 + i;
      const float yn = (ys[i] - mean) * rstd * g2[c] + b2[c];
      __builtin_nontemporal_store(fmaxf(yn + xv[i], 0.f), ob + (size_t)c * HW + p);
    }
  }
}

extern "C" void kernel_launch(void* const* d_in, const int* in_sizes, int n_in,
                              void* d_out, int out_size, void* d_ws, size_t ws_size,
                              hipStream_t stream) {
  (void)in_sizes; (void)n_in; (void)out_size; (void)ws_size;
  const float* x   = (const float*)d_in[0];
  const float* Wq  = (const float*)d_in[1];
  const float* Wk  = (const float*)d_in[2];
  const float* Wv  = (const float*)d_in[3];
  const float* Wo  = (const float*)d_in[4];
  const float* F1w = (const float*)d_in[5];
  const float* F1b = (const float*)d_in[6];
  const float* F2w = (const float*)d_in[7];
  const float* F2b = (const float*)d_in[8];
  const float* g1  = (const float*)d_in[9];
  const float* b1  = (const float*)d_in[10];
  const float* g2  = (const float*)d_in[11];
  const float* b2  = (const float*)d_in[12];
  h16* wsc = (h16*)d_ws;  // 24576 halfs = 48KB of pre-swizzled weight fragments

  tcb_prep_weights<<<96, 256, 0, stream>>>(Wq, Wk, Wv, Wo, F1w, F2w, wsc);

  const int nblocks = (32 * HW) / BLKPIX;                 // 8192
  const size_t shmem = (24576 + WAVES * 16 * 256) * sizeof(h16);  // 80 KB
  tcb_fused<<<nblocks, 128, shmem, stream>>>(x, wsc, g1, b1, g2, b2,
                                             F1b, F2b, (float*)d_out);
}
